// DynamicLinear_67061619359750
// MI455X (gfx1250) — matrix operand order repaired and checked by reading, not compiled, not run on hardware
//
#include <hip/hip_runtime.h>

// ---- CDNA5 (gfx1250) WMMA types -------------------------------------------
typedef __attribute__((ext_vector_type(16))) __bf16   v16bf;
typedef __attribute__((ext_vector_type(8)))  float    v8f;
typedef __attribute__((ext_vector_type(4)))  float    f32x4;
typedef __attribute__((ext_vector_type(4)))  unsigned u32x4;

#define B_SZ   8192
#define IN_SZ  1024
#define OUT_SZ 1024
#define D_SZ   8
#define KTOT   (D_SZ * IN_SZ)   // 8192 fused K  (k = d*1024 + i)
#define BM     128
#define BN     128
#define BK     64               // two 16x16x32 k-steps per staged buffer
#define NKB    (KTOT / BK)      // 128 K-blocks

union Pack32 { __bf16 h[32]; u32x4 q[4]; };   // 32 bf16 = 64B, four 16B chunks
union Frag   { v16bf v;      u32x4 q[2]; };   // WMMA operand = 8 VGPRs/lane

__global__ __launch_bounds__(256)
void dynlin_wmma_bf16(const float* __restrict__ input,
                      const float* __restrict__ w,
                      const float* __restrict__ weights,
                      const float* __restrict__ biases,
                      float* __restrict__ out)
{
    __shared__ __bf16 sA[2][BM][BK];      // 32 KB (double buffered)
    __shared__ __bf16 sB[2][BN][BK];      // 32 KB
    __shared__ float  sW[BM][D_SZ];       //  4 KB : per-row mixing coeffs
    __shared__ float  sBias[D_SZ][BN];    //  4 KB : bias tile

    const int tid  = threadIdx.x;
    const int lane = tid & 31;
    const int wv   = tid >> 5;
    const int wm   = wv & 3;              // 4 waves along M -> 32 rows each
    const int wn   = wv >> 2;             // 2 waves along N -> 64 cols each
    const int bm0  = blockIdx.y * BM;
    const int bn0  = blockIdx.x * BN;

    // ---- prologue: cache w-rows and bias tile in LDS ----
    for (int idx = tid; idx < BM * D_SZ; idx += 256)
        sW[idx / D_SZ][idx % D_SZ] =
            w[(size_t)(bm0 + idx / D_SZ) * D_SZ + (idx % D_SZ)];
    for (int idx = tid; idx < D_SZ * BN; idx += 256)
        sBias[idx / BN][idx % BN] =
            biases[(size_t)(idx / BN) * OUT_SZ + bn0 + (idx % BN)];

    // staging geometry: 2 threads per tile row, 32 floats each
    const int srow = tid >> 1;            // 0..127
    const int scol = (tid & 1) * 32;      // 0 or 32 (element offset within BK)

    v8f acc[2][4];
    #pragma unroll
    for (int i = 0; i < 2; ++i)
        #pragma unroll
        for (int j = 0; j < 4; ++j)
            acc[i][j] = v8f{};

    __syncthreads();                      // sW valid before staging uses it

    // ---- stage K-block 0 into buffer 0 ----
    {
        const float wd = sW[srow][0];     // kb=0 -> d=0, i0=0
        const f32x4* Asrc = (const f32x4*)(input   + (size_t)(bm0 + srow) * IN_SZ + scol);
        const f32x4* Bsrc = (const f32x4*)(weights + (size_t)(bn0 + srow) * IN_SZ + scol);
        Pack32 pa, pb;
        #pragma unroll
        for (int v = 0; v < 8; ++v) {
            f32x4 a = Asrc[v], b = Bsrc[v];
            pa.h[4*v+0] = (__bf16)(a.x * wd); pa.h[4*v+1] = (__bf16)(a.y * wd);
            pa.h[4*v+2] = (__bf16)(a.z * wd); pa.h[4*v+3] = (__bf16)(a.w * wd);
            pb.h[4*v+0] = (__bf16)b.x;        pb.h[4*v+1] = (__bf16)b.y;
            pb.h[4*v+2] = (__bf16)b.z;        pb.h[4*v+3] = (__bf16)b.w;
        }
        #pragma unroll
        for (int c = 0; c < 4; ++c) {
            *(u32x4*)&sA[0][srow][scol + 8 * c] = pa.q[c];
            *(u32x4*)&sB[0][srow][scol + 8 * c] = pb.q[c];
        }
    }
    __syncthreads();

    // ---- main K loop: double buffered, one barrier per 16 WMMAs ----
    int buf = 0;
    for (int kb = 0; kb < NKB; ++kb) {
        if (kb + 1 < NKB) {
            const int kk = (kb + 1) * BK;
            const int d  = kk >> 10;              // 1024 per d-segment (1024 % 64 == 0)
            const int i0 = kk & (IN_SZ - 1);
            const float wd = sW[srow][d];
            const f32x4* Asrc = (const f32x4*)(input   + (size_t)(bm0 + srow) * IN_SZ + i0 + scol);
            const f32x4* Bsrc = (const f32x4*)(weights + ((size_t)d * OUT_SZ + bn0 + srow) * IN_SZ + i0 + scol);
            Pack32 pa, pb;
            #pragma unroll
            for (int v = 0; v < 8; ++v) {
                f32x4 a = Asrc[v], b = Bsrc[v];
                pa.h[4*v+0] = (__bf16)(a.x * wd); pa.h[4*v+1] = (__bf16)(a.y * wd);
                pa.h[4*v+2] = (__bf16)(a.z * wd); pa.h[4*v+3] = (__bf16)(a.w * wd);
                pb.h[4*v+0] = (__bf16)b.x;        pb.h[4*v+1] = (__bf16)b.y;
                pb.h[4*v+2] = (__bf16)b.z;        pb.h[4*v+3] = (__bf16)b.w;
            }
            const int nb = buf ^ 1;               // other buffer: no barrier needed
            #pragma unroll
            for (int c = 0; c < 4; ++c) {
                *(u32x4*)&sA[nb][srow][scol + 8 * c] = pa.q[c];
                *(u32x4*)&sB[nb][srow][scol + 8 * c] = pb.q[c];
            }

            if (kb + 2 < NKB) {                   // warm WGP cache ahead of the double buffer
                const int kk2 = (kb + 2) * BK;
                const int d2  = kk2 >> 10;
                const int i02 = kk2 & (IN_SZ - 1);
                __builtin_prefetch(input   + (size_t)(bm0 + srow) * IN_SZ + i02 + scol, 0, 3);
                __builtin_prefetch(weights + ((size_t)d2 * OUT_SZ + bn0 + srow) * IN_SZ + i02 + scol, 0, 3);
            }
        }

        // two 16x16x32 k-steps from the current buffer
        #pragma unroll
        for (int ks = 0; ks < 2; ++ks) {
            const int frow = lane & 15;
            const int akc  = ks * 32 + (lane >> 4) * 8;   // A: K chunks {base, base+16}
            const int bkc  = ks * 32 + (lane >> 4) * 8;   // B: K chunks {base, base+16}
            Frag fa[2], fb[4];
            #pragma unroll
            for (int tm = 0; tm < 2; ++tm) {
                const __bf16* p = &sA[buf][wm * 32 + tm * 16 + frow][akc];
                fa[tm].q[0] = *(const u32x4*)p;
                fa[tm].q[1] = *(const u32x4*)(p + 16);
            }
            #pragma unroll
            for (int tn = 0; tn < 4; ++tn) {
                const __bf16* p = &sB[buf][wn * 64 + tn * 16 + frow][bkc];
                fb[tn].q[0] = *(const u32x4*)p;
                fb[tn].q[1] = *(const u32x4*)(p + 16);
            }
            #pragma unroll
            for (int tm = 0; tm < 2; ++tm)
                #pragma unroll
                for (int tn = 0; tn < 4; ++tn)
                    acc[tm][tn] = __builtin_amdgcn_wmma_f32_16x16x32_bf16(
                        false, fa[tm].v, false, fb[tn].v,
                        (short)0, acc[tm][tn], false, false);
        }

        __syncthreads();
        buf ^= 1;
    }

    // ---- epilogue: add w @ biases, store f32 ----
    // C/D layout: VGPR r -> M = r + (lane/16)*8, N = lane%16
    #pragma unroll
    for (int tm = 0; tm < 2; ++tm) {
        #pragma unroll
        for (int tn = 0; tn < 4; ++tn) {
            #pragma unroll
            for (int r = 0; r < 8; ++r) {
                const int ml = wm * 32 + tm * 16 + ((lane >> 4) << 3) + r;
                const int nl = wn * 64 + tn * 16 + (lane & 15);
                float bsum = 0.f;
                #pragma unroll
                for (int d = 0; d < D_SZ; ++d)
                    bsum += sW[ml][d] * sBias[d][nl];
                out[(size_t)(bm0 + ml) * OUT_SZ + bn0 + nl] = acc[tm][tn][r] + bsum;
            }
        }
    }
}

extern "C" void kernel_launch(void* const* d_in, const int* in_sizes, int n_in,
                              void* d_out, int out_size, void* d_ws, size_t ws_size,
                              hipStream_t stream) {
    const float* input   = (const float*)d_in[0];   // [B, IN]
    const float* w       = (const float*)d_in[1];   // [B, D]
    const float* weights = (const float*)d_in[2];   // [D, OUT, IN]
    const float* biases  = (const float*)d_in[3];   // [D, OUT]
    float* out = (float*)d_out;                     // [B, OUT]

    dim3 grid(OUT_SZ / BN, B_SZ / BM);              // (8, 64)
    dynlin_wmma_bf16<<<grid, 256, 0, stream>>>(input, w, weights, biases, out);
}
